// MultiHeadAttention_33621003993262
// MI455X (gfx1250) — matrix operand order repaired and checked by reading, not compiled, not run on hardware
//
#include <hip/hip_runtime.h>

typedef __attribute__((ext_vector_type(16))) __bf16 v16bf;
typedef __attribute__((ext_vector_type(8)))  float  v8f;
typedef __attribute__((ext_vector_type(4)))  int    i4;
typedef __attribute__((ext_vector_type(4)))  float  f4;
typedef __attribute__((ext_vector_type(4)))  unsigned short us4;

#ifndef __has_builtin
#define __has_builtin(x) 0
#endif

#if __has_builtin(__builtin_amdgcn_global_load_async_to_lds_b128)
#define ASYNC_LDS 1
typedef __attribute__((address_space(1))) i4 as1_i4;  // global v4i32
typedef __attribute__((address_space(3))) i4 as3_i4;  // LDS v4i32
#else
#define ASYNC_LDS 0
#endif

union FragAB {
  i4    i[2];   // 32 bytes = 16 bf16
  v16bf v;
};

__device__ __forceinline__ unsigned short f2bf(float f) {
  union { float f; unsigned u; } x;
  x.f = f;
  unsigned u = x.u;
  return (unsigned short)((u + 0x7FFFu + ((u >> 16) & 1u)) >> 16);
}

__device__ __forceinline__ v8f wmma_bf16(const FragAB& a, const FragAB& b, v8f c) {
  // D = A(16x32 bf16) * B(32x16 bf16) + C(16x16 f32)
  return __builtin_amdgcn_wmma_f32_16x16x32_bf16(
      /*neg_a=*/false, a.v, /*neg_b=*/false, b.v,
      /*c_mod=*/(short)0, c, /*reuse_a=*/false, /*reuse_b=*/false);
}

// 16-byte global -> LDS copy: async (ASYNCcnt / dedicated DMA path) when the
// toolchain exposes the gfx1250 builtin, else VGPR bounce.
__device__ __forceinline__ void cp16(const unsigned short* g, unsigned short* l) {
#if ASYNC_LDS
  __builtin_amdgcn_global_load_async_to_lds_b128(
      (as1_i4*)g, (as3_i4*)l, 0, 0);
#else
  *(i4*)l = *(const i4*)g;
#endif
}

__device__ __forceinline__ void cp_wait() {
#if ASYNC_LDS
#if __has_builtin(__builtin_amdgcn_s_wait_asynccnt)
  __builtin_amdgcn_s_wait_asynccnt(0);
#else
  asm volatile("s_wait_asynccnt 0x0" ::: "memory");
#endif
#endif
}

// ---------------------------------------------------------------------------
// fp32 -> bf16 conversion (4 elements / thread)
// ---------------------------------------------------------------------------
__global__ __launch_bounds__(256) void cvt_f32_bf16(const float* __restrict__ src,
                                                    unsigned short* __restrict__ dst,
                                                    int n4) {
  int i = blockIdx.x * 256 + threadIdx.x;
  if (i >= n4) return;
  f4 f = ((const f4*)src)[i];
  us4 o;
  o.x = f2bf(f.x); o.y = f2bf(f.y); o.z = f2bf(f.z); o.w = f2bf(f.w);
  ((us4*)dst)[i] = o;
}

// ---------------------------------------------------------------------------
// bf16 GEMM:  Y[M,1024] = X[M,1024] @ W[1024,1024] + bias
// Block tile 128x64, 8 waves, wave tile 32x32 (2x2 wmma), K-step 32.
// Double-buffered LDS + async global->LDS staging: one barrier per K-step.
// mode 0: fp32 row-major out          (final projection)
// mode 1: bf16 out, [B,H,S,64] layout (Q and V projections)
// mode 2: bf16 out, [B,H,64,S] layout (K projection, transposed)
// ---------------------------------------------------------------------------
__global__ __launch_bounds__(256) void gemm_bf16(const unsigned short* __restrict__ X,
                                                 const unsigned short* __restrict__ W,
                                                 const float* __restrict__ bias,
                                                 void* __restrict__ out,
                                                 int mode) {
  __shared__ unsigned short As[2][128 * 40];  // 128 rows x 32 bf16 (+8 pad)
  __shared__ unsigned short Bs[2][32 * 72];   // 32 rows  x 64 bf16 (+8 pad)

  const int K = 1024, N = 1024;
  const int tid  = threadIdx.x;
  const int lane = tid & 31;
  const int wid  = tid >> 5;
  const int wm = wid >> 1, wn = wid & 1;
  const int mbase = blockIdx.x * 128;
  const int nbase = blockIdx.y * 64;

  v8f acc[2][2] = {};

  const int a_row = tid >> 1, a_half = tid & 1;  // A tile loader: 32B/thread
  const int b_row = tid >> 3, b_seg  = tid & 7;  // B tile loader: 16B/thread
  const int klo = (lane < 16) ? 0 : 8;           // A-fragment K offset per lane half

  auto load_tile = [&](int buf, int k0) {
    const unsigned short* ga = X + (size_t)(mbase + a_row) * K + k0 + a_half * 16;
    unsigned short* la = &As[buf][a_row * 40 + a_half * 16];
    cp16(ga, la);
    cp16(ga + 8, la + 8);
    cp16(W + (size_t)(k0 + b_row) * N + nbase + b_seg * 8,
         &Bs[buf][b_row * 72 + b_seg * 8]);
  };

  load_tile(0, 0);

  for (int k0 = 0; k0 < K; k0 += 32) {
    const int cur = (k0 >> 5) & 1;
    cp_wait();        // own async stores to LDS complete
    __syncthreads();  // tile `cur` visible to all waves; buffer cur^1 free
    if (k0 + 32 < K) load_tile(cur ^ 1, k0 + 32);

    FragAB a[2], b[2];
#pragma unroll
    for (int mi = 0; mi < 2; ++mi) {
      const unsigned short* p = As[cur] + (wm * 32 + mi * 16 + (lane & 15)) * 40;
      a[mi].i[0] = *(const i4*)(p + klo);        // K {0..7} / {8..15}
      a[mi].i[1] = *(const i4*)(p + klo + 16);   // K {16..23} / {24..31}
    }
#pragma unroll
    for (int ni = 0; ni < 2; ++ni) {
      const unsigned short* p = Bs[cur] + wn * 32 + ni * 16 + (lane & 15);  // lane&15 = N col
#pragma unroll
      for (int s = 0; s < 16; ++s)
        b[ni].v[s] = __builtin_bit_cast(__bf16, p[(klo + (s & 7) + ((s >> 3) << 4)) * 72]);
    }
#pragma unroll
    for (int mi = 0; mi < 2; ++mi)
#pragma unroll
      for (int ni = 0; ni < 2; ++ni)
        acc[mi][ni] = wmma_bf16(a[mi], b[ni], acc[mi][ni]);
  }

  // epilogue: C layout -> vgpr g: lanes 0-15 row g, lanes 16-31 row g+8; col = lane&15
  const int mrow_hi = (lane >> 4) << 3;
#pragma unroll
  for (int mi = 0; mi < 2; ++mi)
#pragma unroll
    for (int ni = 0; ni < 2; ++ni) {
      const int ncol = nbase + wn * 32 + ni * 16 + (lane & 15);
      const float bv = bias[ncol];
#pragma unroll
      for (int g = 0; g < 8; ++g) {
        const int mrow = mbase + wm * 32 + mi * 16 + g + mrow_hi;
        const float v = acc[mi][ni][g] + bv;
        if (mode == 0) {
          ((float*)out)[(size_t)mrow * N + ncol] = v;
        } else {
          const int bb = mrow >> 11, s = mrow & 2047;
          const int h = ncol >> 6, d = ncol & 63;
          const size_t idx = (mode == 1)
              ? ((size_t)((bb * 16 + h) * 2048 + s) * 64 + d)
              : ((size_t)((bb * 16 + h) * 64 + d) * 2048 + s);
          ((unsigned short*)out)[idx] = f2bf(v);
        }
      }
    }
}

// ---------------------------------------------------------------------------
// Flash attention: one wave per 16-query tile, 8 tiles (128 q) per block.
// K/V staged in double-buffered LDS via async copies; online softmax.
// Qh: [BH, S, 64]  Kt: [BH, 64, S]  Vh: [BH, S, 64]  ctx: [B, S, 1024] bf16
// ---------------------------------------------------------------------------
__global__ __launch_bounds__(256) void flash_attn(const unsigned short* __restrict__ Qh,
                                                  const unsigned short* __restrict__ Kt,
                                                  const unsigned short* __restrict__ Vh,
                                                  unsigned short* __restrict__ ctx) {
  __shared__ unsigned short Ks[2][64 * 40];    // 64 d-rows x 32 kv (+8 pad)
  __shared__ unsigned short Vs[2][32 * 72];    // 32 kv-rows x 64 d (+8 pad)
  __shared__ unsigned short Ps[8 * 16 * 40];   // per-wave P scratch 16q x 32kv (+8 pad)

  const int S = 2048;
  const int tid = threadIdx.x, lane = tid & 31, wid = tid >> 5;
  const int bh = blockIdx.y;
  const int b = bh >> 4, h = bh & 15;
  const int q0 = blockIdx.x * 128 + wid * 16;

  const unsigned short* Qg = Qh + (size_t)bh * S * 64;
  const unsigned short* Kg = Kt + (size_t)bh * 64 * S;
  const unsigned short* Vg = Vh + (size_t)bh * S * 64;

  // Q fragments (K-dim = d = 64 -> 2 fragments), held in registers
  FragAB qf[2];
  {
    const int row = q0 + (lane & 15);
    const int lo = (lane < 16) ? 0 : 8;
#pragma unroll
    for (int f = 0; f < 2; ++f) {
      const unsigned short* p = Qg + (size_t)row * 64 + f * 32 + lo;
      qf[f].i[0] = *(const i4*)p;
      qf[f].i[1] = *(const i4*)(p + 16);
    }
  }

  v8f acco[4] = {};
  float m[8], l[8];
#pragma unroll
  for (int g = 0; g < 8; ++g) { m[g] = -1e30f; l[g] = 0.f; }
  const float scale = 0.125f;  // 1/sqrt(64)

  const int k_d = tid >> 2, k_seg = tid & 3;  // K-tile loader (16B/thread)
  const int v_r = tid >> 3, v_seg = tid & 7;  // V-tile loader (16B/thread)
  unsigned short* Pw = Ps + wid * 16 * 40;
  const int klo = (lane < 16) ? 0 : 8;

  auto load_kv = [&](int buf, int kv0) {
    cp16(Kg + (size_t)k_d * S + kv0 + k_seg * 8, &Ks[buf][k_d * 40 + k_seg * 8]);
    cp16(Vg + (size_t)(kv0 + v_r) * 64 + v_seg * 8, &Vs[buf][v_r * 72 + v_seg * 8]);
  };

  load_kv(0, 0);

  for (int kv0 = 0; kv0 < S; kv0 += 32) {
    const int cur = (kv0 >> 5) & 1;
    cp_wait();        // own async stores complete
    __syncthreads();  // tile `cur` ready; buffer cur^1 free
    if (kv0 + 32 < S) load_kv(cur ^ 1, kv0 + 32);

    // scores S(16q x 32kv) = Q @ K^T  (4 wmma: 2 K-steps over d, 2 kv halves)
    v8f sc[2] = {};
#pragma unroll
    for (int ks = 0; ks < 2; ++ks) {
      FragAB bk[2];
#pragma unroll
      for (int ni = 0; ni < 2; ++ni) {
        const unsigned short* p = Ks[cur] + (ks * 32) * 40 + ni * 16 + (lane & 15);  // lane&15 = kv col
#pragma unroll
        for (int s = 0; s < 16; ++s)
          bk[ni].v[s] = __builtin_bit_cast(__bf16, p[(klo + (s & 7) + ((s >> 3) << 4)) * 40]);
      }
      sc[0] = wmma_bf16(qf[ks], bk[0], sc[0]);
      sc[1] = wmma_bf16(qf[ks], bk[1], sc[1]);
    }

    // online softmax; row statistics live replicated across each 16-lane group
    const int col = lane & 15;
#pragma unroll
    for (int g = 0; g < 8; ++g) {
      float s0 = sc[0][g] * scale, s1 = sc[1][g] * scale;
      float rm = fmaxf(s0, s1);
#pragma unroll
      for (int off = 8; off >= 1; off >>= 1) rm = fmaxf(rm, __shfl_xor(rm, off));
      const float mn = fmaxf(m[g], rm);
      const float alpha = __expf(m[g] - mn);
      m[g] = mn;
      const float p0 = __expf(s0 - mn), p1 = __expf(s1 - mn);
      float rs = p0 + p1;
#pragma unroll
      for (int off = 8; off >= 1; off >>= 1) rs += __shfl_xor(rs, off);
      l[g] = l[g] * alpha + rs;
#pragma unroll
      for (int ni = 0; ni < 4; ++ni) acco[ni][g] *= alpha;
      // C-layout -> LDS (wave-private; DS ops are in-order within a wave)
      const int r = g + ((lane >> 4) << 3);
      Pw[r * 40 + col]      = f2bf(p0);
      Pw[r * 40 + col + 16] = f2bf(p1);
    }

    // context += P(16x32) @ V(32x64): 1 A-frag, 4 d-tiles
    FragAB pa;
    {
      const unsigned short* p = Pw + (lane & 15) * 40;
      pa.i[0] = *(const i4*)(p + klo);
      pa.i[1] = *(const i4*)(p + klo + 16);
    }
#pragma unroll
    for (int ni = 0; ni < 4; ++ni) {
      FragAB bv;
      const unsigned short* p = Vs[cur] + ni * 16 + (lane & 15);  // lane&15 = d col
#pragma unroll
      for (int s = 0; s < 16; ++s)
        bv.v[s] = __builtin_bit_cast(__bf16, p[(klo + (s & 7) + ((s >> 3) << 4)) * 72]);
      acco[ni] = wmma_bf16(pa, bv, acco[ni]);
    }
  }

  // normalize and write ctx as bf16 [B,S,1024]
#pragma unroll
  for (int ni = 0; ni < 4; ++ni) {
    const int colD = h * 64 + ni * 16 + (lane & 15);
#pragma unroll
    for (int g = 0; g < 8; ++g) {
      const int q = q0 + g + ((lane >> 4) << 3);
      const float v = acco[ni][g] / l[g];
      ctx[(size_t)(b * S + q) * 1024 + colD] = f2bf(v);
    }
  }
}

// ---------------------------------------------------------------------------
// Host launcher
// ---------------------------------------------------------------------------
extern "C" void kernel_launch(void* const* d_in, const int* in_sizes, int n_in,
                              void* d_out, int out_size, void* d_ws, size_t ws_size,
                              hipStream_t stream) {
  const float* q     = (const float*)d_in[0];
  const float* k     = (const float*)d_in[1];
  const float* v     = (const float*)d_in[2];
  const float* w_q   = (const float*)d_in[3];
  const float* b_q   = (const float*)d_in[4];
  const float* w_k   = (const float*)d_in[5];
  const float* b_k   = (const float*)d_in[6];
  const float* w_v   = (const float*)d_in[7];
  const float* b_v   = (const float*)d_in[8];
  const float* w_out = (const float*)d_in[9];
  const float* b_out = (const float*)d_in[10];

  const size_t nx = (size_t)2 * 2048 * 1024;  // 4,194,304 (q/k/v elements)
  const size_t nw = (size_t)1024 * 1024;      // 1,048,576 (weight elements)

  char* ws = (char*)d_ws;
  unsigned short* qb  = (unsigned short*)ws; ws += nx * 2;
  unsigned short* kb  = (unsigned short*)ws; ws += nx * 2;
  unsigned short* vb  = (unsigned short*)ws; ws += nx * 2;
  unsigned short* wqb = (unsigned short*)ws; ws += nw * 2;
  unsigned short* wkb = (unsigned short*)ws; ws += nw * 2;
  unsigned short* wvb = (unsigned short*)ws; ws += nw * 2;
  unsigned short* wob = (unsigned short*)ws; ws += nw * 2;
  unsigned short* Qh  = (unsigned short*)ws; ws += nx * 2;  // [BH,S,64]
  unsigned short* Kts = (unsigned short*)ws; ws += nx * 2;  // [BH,64,S]
  unsigned short* Vh  = (unsigned short*)ws; ws += nx * 2;  // [BH,S,64]
  unsigned short* ctx = (unsigned short*)ws; ws += nx * 2;  // [B,S,1024]

  cvt_f32_bf16<<<(int)(nx / 1024), 256, 0, stream>>>(q, qb, (int)(nx / 4));
  cvt_f32_bf16<<<(int)(nx / 1024), 256, 0, stream>>>(k, kb, (int)(nx / 4));
  cvt_f32_bf16<<<(int)(nx / 1024), 256, 0, stream>>>(v, vb, (int)(nx / 4));
  cvt_f32_bf16<<<(int)(nw / 1024), 256, 0, stream>>>(w_q, wqb, (int)(nw / 4));
  cvt_f32_bf16<<<(int)(nw / 1024), 256, 0, stream>>>(w_k, wkb, (int)(nw / 4));
  cvt_f32_bf16<<<(int)(nw / 1024), 256, 0, stream>>>(w_v, wvb, (int)(nw / 4));
  cvt_f32_bf16<<<(int)(nw / 1024), 256, 0, stream>>>(w_out, wob, (int)(nw / 4));

  dim3 gg(32, 16), bb(256);
  gemm_bf16<<<gg, bb, 0, stream>>>(qb, wqb, b_q, Qh, 1);   // Q -> [BH,S,64]
  gemm_bf16<<<gg, bb, 0, stream>>>(kb, wkb, b_k, Kts, 2);  // K -> [BH,64,S]
  gemm_bf16<<<gg, bb, 0, stream>>>(vb, wvb, b_v, Vh, 1);   // V -> [BH,S,64]

  flash_attn<<<dim3(16, 32), 256, 0, stream>>>(Qh, Kts, Vh, ctx);

  gemm_bf16<<<gg, bb, 0, stream>>>(ctx, wob, b_out, d_out, 0);  // fp32 out
}
